// VqVae_80504866996931
// MI455X (gfx1250) — matrix operand order repaired and checked by reading, not compiled, not run on hardware
//
#include <hip/hip_runtime.h>
#include <hip/hip_bf16.h>

typedef unsigned short u16;
typedef unsigned int   u32;
typedef __attribute__((ext_vector_type(16))) __bf16 v16bf;
typedef __attribute__((ext_vector_type(8)))  float  v8f;
typedef __attribute__((ext_vector_type(4)))  u32    u32x4;

__device__ __forceinline__ u16 f2bf(float f) {
  union { float f; u32 u; } v; v.f = f;
  u32 r = v.u + 0x7FFFu + ((v.u >> 16) & 1u);
  return (u16)(r >> 16);
}
__device__ __forceinline__ float bf2f(u16 h) {
  union { u32 u; float f; } v; v.u = ((u32)h) << 16; return v.f;
}
__device__ __forceinline__ u32 relu2(u32 v) {
  u32 lo = (v & 0x8000u)     ? 0u : (v & 0xFFFFu);
  u32 hi = (v & 0x80000000u) ? 0u : (v & 0xFFFF0000u);
  return lo | hi;
}

// ---------------------------------------------------------------------------
// Direct conv for encoder layer 1 (Cin=3, K=48 -> not WMMA-friendly).
// x: f32 NCHW (32,3,128,128), w: OIHW (256,3,4,4), out: bf16 NHWC (32,64,64,256)
// ---------------------------------------------------------------------------
__global__ __launch_bounds__(256) void enc_conv1_kernel(
    const float* __restrict__ x, const float* __restrict__ w,
    const float* __restrict__ b, u16* __restrict__ out)
{
  int t = blockIdx.x * 256 + threadIdx.x;   // (n,oh,ow,co)
  int co = t & 255; int t2 = t >> 8;
  int ow = t2 & 63; t2 >>= 6;
  int oh = t2 & 63; int n = t2 >> 6;
  float acc = b[co];
  #pragma unroll
  for (int ci = 0; ci < 3; ++ci) {
    #pragma unroll
    for (int kh = 0; kh < 4; ++kh) {
      int ih = oh * 2 - 1 + kh;
      if (ih < 0 || ih >= 128) continue;
      #pragma unroll
      for (int kw = 0; kw < 4; ++kw) {
        int iw = ow * 2 - 1 + kw;
        if (iw < 0 || iw >= 128) continue;
        acc += x[((n * 3 + ci) * 128 + ih) * 128 + iw] *
               w[((co * 3 + ci) * 4 + kh) * 4 + kw];
      }
    }
  }
  out[t] = f2bf(acc);
}

// ---------------------------------------------------------------------------
// Implicit-GEMM conv / conv-transpose-parity-class kernel, WMMA bf16.
//   out[m][co] = sum_k A[m][k] * W[co][k] + bias[co]  (+ optional residual)
// Block tile 128(M) x 64(N), K-step 32, double-buffered LDS (1 barrier/step,
// global loads for step ks+1 overlap the 4 WMMAs of step ks).
// 8 waves (4x2), each wave 32x32 -> 4 accumulators, 2x A/B fragment reuse.
// All spatial dims are powers of two: owsh/ohsh are log2(OW)/log2(OH).
// mode 0: conv (KH*KW taps, stride, pad). mode 1: convtranspose k4 s2 p1
//         parity class opq (op=oh&1, oq=ow&1), K = 4 taps x Cin.
// f32_mode: 0 none, 1 row-major [M][CoutReal] f32, 2 NCHW f32.
// ---------------------------------------------------------------------------
__global__ __launch_bounds__(256) void gemm_conv_wmma_kernel(
    const u16* __restrict__ in, const u16* __restrict__ wpack,
    const float* __restrict__ bias, const u16* __restrict__ resid,
    u16* __restrict__ outBf, float* __restrict__ outF32,
    int IH, int IW, int Cin,
    int OH, int OW, int owsh, int ohsh,
    int CoutPad, int CoutReal,
    int KW, int stride, int pad, int Ktot,
    int mode, int opq, int relu_in, int f32_mode)
{
  __shared__ u32 Als[2][128][20];   // 2 x (128 rows x 32 bf16 (+pad))
  __shared__ u32 Bls[2][64][20];    // 2 x (64 cols x 32 bf16 (+pad))

  const int tid  = threadIdx.x;
  const int wave = tid >> 5, lane = tid & 31;
  const int wm = wave >> 1, wn = wave & 1;
  const int hf = lane >> 4, lr = lane & 15;
  const int m0  = blockIdx.x * 128;
  const int co0 = blockIdx.y * 64;
  const int op = opq >> 1, oq = opq & 1;

  // --- A staging: 512 16B chunks; each thread owns rows tid>>2 and 64+(tid>>2)
  const int achunk = tid & 3;
  int arow[2];  arow[0] = tid >> 2;  arow[1] = 64 + (tid >> 2);
  int aoh[2], aow[2];         // mode0: (oh,ow); mode1: (oh2,ow2)
  long inBase[2];
  #pragma unroll
  for (int i = 0; i < 2; ++i) {
    int m = m0 + arow[i];
    int nn;
    if (mode == 0) {
      aow[i] = m & (OW - 1); aoh[i] = (m >> owsh) & (OH - 1); nn = m >> (owsh + ohsh);
    } else {
      int ws = owsh - 1, hs = ohsh - 1;
      aow[i] = m & ((OW >> 1) - 1); aoh[i] = (m >> ws) & ((OH >> 1) - 1); nn = m >> (ws + hs);
    }
    inBase[i] = (long)nn * IH * IW * Cin;
  }
  const int bcol = tid >> 2, bchunk = tid & 3;   // 64 cols x 4 chunks = 256

  v8f acc[2][2];
  #pragma unroll
  for (int mi = 0; mi < 2; ++mi)
    #pragma unroll
    for (int ni = 0; ni < 2; ++ni)
      acc[mi][ni] = (v8f){0.f,0.f,0.f,0.f,0.f,0.f,0.f,0.f};

  int kh = 0, kw = 0, tap = 0, ci0 = 0;
  const int nK = Ktot >> 5;

  auto gather = [&](int ksIdx, u32x4 av[2], u32x4& bv) {
    #pragma unroll
    for (int i = 0; i < 2; ++i) {
      int ih, iw;
      if (mode == 0) { ih = aoh[i] * stride - pad + kh; iw = aow[i] * stride - pad + kw; }
      else           { ih = aoh[i] + op - (tap >> 1);   iw = aow[i] + oq - (tap & 1);   }
      av[i] = (u32x4){0u, 0u, 0u, 0u};
      if (ih >= 0 && ih < IH && iw >= 0 && iw < IW) {
        av[i] = *(const u32x4*)(in + inBase[i] + ((long)ih * IW + iw) * Cin
                                + ci0 + achunk * 8);
        if (relu_in) {
          av[i][0]=relu2(av[i][0]); av[i][1]=relu2(av[i][1]);
          av[i][2]=relu2(av[i][2]); av[i][3]=relu2(av[i][3]);
        }
      }
    }
    bv = *(const u32x4*)(wpack + (long)(co0 + bcol) * Ktot + (ksIdx << 5) + bchunk * 8);
  };
  auto advanceK = [&]() {
    ci0 += 32;
    if (ci0 == Cin) { ci0 = 0; ++tap; ++kw; if (kw == KW) { kw = 0; ++kh; } }
  };
  auto stage = [&](int buf, const u32x4 av[2], const u32x4& bv) {
    #pragma unroll
    for (int i = 0; i < 2; ++i) {
      u32* d = &Als[buf][arow[i]][achunk * 4];
      d[0]=av[i][0]; d[1]=av[i][1]; d[2]=av[i][2]; d[3]=av[i][3];
    }
    u32* d = &Bls[buf][bcol][bchunk * 4];
    d[0]=bv[0]; d[1]=bv[1]; d[2]=bv[2]; d[3]=bv[3];
  };

  // prologue: stage K-step 0 into buffer 0
  { u32x4 av[2], bv; gather(0, av, bv); stage(0, av, bv); advanceK(); }
  __syncthreads();

  for (int ks = 0; ks < nK; ++ks) {
    const int cur = ks & 1;
    const bool have = (ks + 1) < nK;       // uniform across block
    u32x4 av[2], bv;
    if (have) {
      gather(ks + 1, av, bv);              // loads in flight during WMMAs
      if (ks + 2 < nK)                     // gfx1250 global_prefetch_b8
        __builtin_prefetch(wpack + (long)(co0 + bcol) * Ktot + ((ks + 2) << 5)
                           + bchunk * 8, 0, 0);
    }

    // ---- fragments per ISA 16-bit layouts, from buffer `cur` ----
    union { v16bf v; u32 u[8]; } fa[2], fb[2];
    #pragma unroll
    for (int mi = 0; mi < 2; ++mi) {
      const u32* ar = Als[cur][wm * 32 + mi * 16 + lr];
      #pragma unroll
      for (int j = 0; j < 4; ++j) { fa[mi].u[j] = ar[hf*4 + j]; fa[mi].u[4+j] = ar[8 + hf*4 + j]; }
    }
    #pragma unroll
    for (int ni = 0; ni < 2; ++ni) {
      const u32* br = Bls[cur][wn * 32 + ni * 16 + lr];
      #pragma unroll
      for (int j = 0; j < 4; ++j) { fb[ni].u[j] = br[hf*4 + j]; fb[ni].u[4+j] = br[8 + hf*4 + j]; }
    }
    #pragma unroll
    for (int mi = 0; mi < 2; ++mi)
      #pragma unroll
      for (int ni = 0; ni < 2; ++ni)
        acc[mi][ni] = __builtin_amdgcn_wmma_f32_16x16x32_bf16(
            false, fa[mi].v, false, fb[ni].v, (short)0, acc[mi][ni], false, false);

    if (have) { stage(cur ^ 1, av, bv); advanceK(); }
    __syncthreads();
  }

  // ---- epilogue: bias (+residual), stores ----
  #pragma unroll
  for (int ni = 0; ni < 2; ++ni) {
    const int co = co0 + wn * 32 + ni * 16 + lr;
    const float bval = (co < CoutReal) ? bias[co] : 0.f;
    #pragma unroll
    for (int mi = 0; mi < 2; ++mi) {
      #pragma unroll
      for (int r = 0; r < 8; ++r) {
        int m = m0 + wm * 32 + mi * 16 + 8 * hf + r;
        float v = acc[mi][ni][r] + bval;
        int nn, ohh, oww;
        if (mode == 0) {
          oww = m & (OW - 1); ohh = (m >> owsh) & (OH - 1); nn = m >> (owsh + ohsh);
        } else {
          int ws = owsh - 1, hs = ohsh - 1;
          int w2 = m & ((OW >> 1) - 1); int h2 = (m >> ws) & ((OH >> 1) - 1);
          nn = m >> (ws + hs);
          ohh = 2 * h2 + op; oww = 2 * w2 + oq;
        }
        if (co < CoutReal) {
          long oidx = ((long)(nn * OH + ohh) * OW + oww);
          if (resid)          v += bf2f(resid[oidx * CoutReal + co]);
          if (outBf)          outBf[oidx * CoutReal + co] = f2bf(v);
          if (f32_mode == 1)      outF32[oidx * CoutReal + co] = v;
          else if (f32_mode == 2) outF32[(((long)nn * CoutReal + co) * OH + ohh) * OW + oww] = v;
        }
      }
    }
  }
}

// ---------------------------------------------------------------------------
// Weight prepack: OIHW f32 -> [Cout][(kh*KW+kw)*Cin+ci] bf16
// ---------------------------------------------------------------------------
__global__ void pack_conv_w_kernel(const float* __restrict__ src, u16* __restrict__ dst,
                                   int Cout, int Cin, int KH, int KW)
{
  long total = (long)Cout * KH * KW * Cin;
  long t = (long)blockIdx.x * 256 + threadIdx.x;
  if (t >= total) return;
  int Kt = KH * KW * Cin;
  int co = (int)(t / Kt); int rem = (int)(t % Kt);
  int kh = rem / (KW * Cin); int r2 = rem % (KW * Cin);
  int kw = r2 / Cin; int ci = r2 % Cin;
  dst[t] = f2bf(src[((long)(co * Cin + ci) * KH + kh) * KW + kw]);
}

// ConvTranspose weight (Cin,Cout,4,4) -> 4 parity-class packs [CoutPad][4*Cin]
__global__ void pack_tconv_w_kernel(const float* __restrict__ src, u16* __restrict__ dst,
                                    int CoutReal, int CoutPad, int Cin)
{
  long total = 4L * CoutPad * 4 * Cin;
  long t = (long)blockIdx.x * 256 + threadIdx.x;
  if (t >= total) return;
  int perClass = CoutPad * 4 * Cin;
  int cls = (int)(t / perClass); int rem = (int)(t % perClass);
  int co = rem / (4 * Cin); int r2 = rem % (4 * Cin);
  int tap = r2 / Cin; int ci = r2 % Cin;
  int khi = tap >> 1, kwi = tap & 1;
  int op = cls >> 1, oq = cls & 1;
  int kh = 1 - op + 2 * khi, kw = 1 - oq + 2 * kwi;
  float v = 0.f;
  if (co < CoutReal) v = src[((long)(ci * CoutReal + co) * 4 + kh) * 4 + kw];
  dst[t] = f2bf(v);
}

__global__ void f32_to_bf16_kernel(const float* __restrict__ src, u16* __restrict__ dst, long n)
{
  long t = (long)blockIdx.x * 256 + threadIdx.x;
  if (t < n) dst[t] = f2bf(src[t]);
}

__global__ void code_norm_kernel(const float* __restrict__ codes, float* __restrict__ cn)
{
  int j = blockIdx.x * 256 + threadIdx.x;
  if (j >= 512) return;
  float s = 0.f;
  for (int d = 0; d < 256; ++d) { float v = codes[j * 256 + d]; s += v * v; }
  cn[j] = -0.5f * s;   // argmin ||z-c||^2 == argmax (z.c - 0.5||c||^2)
}

// (32, C=256, 32, 32): bf16 NHWC -> f32 NCHW  (all dims pow2 -> shifts)
__global__ void nhwc_bf16_to_nchw_f32_kernel(const u16* __restrict__ src, float* __restrict__ dst)
{
  long t = (long)blockIdx.x * 256 + threadIdx.x;
  int w = (int)(t & 31); long t2 = t >> 5;
  int h = (int)(t2 & 31); t2 >>= 5;
  int c = (int)(t2 & 255); int n = (int)(t2 >> 8);
  dst[t] = bf2f(src[(((long)n * 32 + h) * 32 + w) * 256 + c]);
}

// One wave per row: argmax over K scores, then gather codebook entry.
__global__ __launch_bounds__(256) void vq_argmin_gather_kernel(
    const float* __restrict__ scores, const float* __restrict__ codes,
    float* __restrict__ zdec_nchw, u16* __restrict__ zq_bf,
    int H, int W, int C, int K)
{
  int wave = threadIdx.x >> 5, lane = threadIdx.x & 31;
  int m = blockIdx.x * 8 + wave;
  const float* row = scores + (long)m * K;
  float bs = -3.4e38f; int bi = 0;
  for (int j = lane; j < K; j += 32) {
    float s = row[j];
    if (s > bs) { bs = s; bi = j; }
  }
  for (int off = 16; off >= 1; off >>= 1) {
    float os = __shfl_xor(bs, off, 32);
    int   oi = __shfl_xor(bi, off, 32);
    if (os > bs || (os == bs && oi < bi)) { bs = os; bi = oi; }
  }
  int w = m & (W - 1); int t2 = m / W; int h = t2 & (H - 1); int n = t2 / H;
  for (int c = lane; c < C; c += 32) {
    float v = codes[(long)bi * C + c];
    zdec_nchw[(((long)n * C + c) * H + h) * W + w] = v;
    zq_bf[(long)m * C + c] = f2bf(v);
  }
}

// ---------------------------------------------------------------------------
extern "C" void kernel_launch(void* const* d_in, const int* in_sizes, int n_in,
                              void* d_out, int out_size, void* d_ws, size_t ws_size,
                              hipStream_t stream)
{
  (void)in_sizes; (void)n_in; (void)out_size; (void)ws_size;
  const float* x      = (const float*)d_in[0];
  const float* codes  = (const float*)d_in[1];
  const float* enc_w1 = (const float*)d_in[2];
  const float* enc_b1 = (const float*)d_in[3];
  const float* enc_w2 = (const float*)d_in[4];
  const float* enc_b2 = (const float*)d_in[5];
  const float* rw1[4]; const float* rb1[4]; const float* rw2[4]; const float* rb2[4];
  for (int i = 0; i < 4; ++i) {
    rw1[i] = (const float*)d_in[6 + i * 4 + 0];
    rb1[i] = (const float*)d_in[6 + i * 4 + 1];
    rw2[i] = (const float*)d_in[6 + i * 4 + 2];
    rb2[i] = (const float*)d_in[6 + i * 4 + 3];
  }
  const float* dec_tw1 = (const float*)d_in[22];
  const float* dec_tb1 = (const float*)d_in[23];
  const float* dec_tw2 = (const float*)d_in[24];
  const float* dec_tb2 = (const float*)d_in[25];

  float* xhat = (float*)d_out;                   // (32,3,128,128)
  float* zout = xhat + 32L * 3 * 128 * 128;      // (32,256,32,32)
  float* zdec = zout + 32L * 256 * 32 * 32;      // (32,256,32,32)

  char* wp = (char*)d_ws;
  auto carve = [&](size_t bytes) { char* p = wp; wp += (bytes + 255) & ~(size_t)255; return p; };
  // 67MB region time-shared: enc1 activation -> VQ scores -> decoder 64x64 act.
  u16*   big    = (u16*)carve(32L * 64 * 64 * 256 * 2);
  float* scores = (float*)big;
  u16* t0 = (u16*)carve(32L * 32 * 32 * 256 * 2);
  u16* t1 = (u16*)carve(32L * 32 * 32 * 256 * 2);
  u16* t2 = (u16*)carve(32L * 32 * 32 * 256 * 2);
  float* cn    = (float*)carve(512 * 4);
  u16* p_enc2  = (u16*)carve(256L * 4096 * 2);
  u16* p_rw1[4]; u16* p_rw2[4];
  for (int i = 0; i < 4; ++i) { p_rw1[i] = (u16*)carve(256L * 2304 * 2); p_rw2[i] = (u16*)carve(256L * 256 * 2); }
  u16* p_codes = (u16*)carve(512L * 256 * 2);
  u16* p_tc1   = (u16*)carve(4L * 256 * 1024 * 2);
  u16* p_tc2   = (u16*)carve(4L * 64 * 1024 * 2);

  dim3 B(256);
  // --- weight prepacks ---
  pack_conv_w_kernel<<<(256 * 4096 + 255) / 256, B, 0, stream>>>(enc_w2, p_enc2, 256, 256, 4, 4);
  for (int i = 0; i < 4; ++i) {
    pack_conv_w_kernel<<<(256 * 2304 + 255) / 256, B, 0, stream>>>(rw1[i], p_rw1[i], 256, 256, 3, 3);
    pack_conv_w_kernel<<<(256 * 256 + 255) / 256, B, 0, stream>>>(rw2[i], p_rw2[i], 256, 256, 1, 1);
  }
  f32_to_bf16_kernel<<<(512 * 256 + 255) / 256, B, 0, stream>>>(codes, p_codes, 512L * 256);
  code_norm_kernel<<<2, B, 0, stream>>>(codes, cn);
  pack_tconv_w_kernel<<<(4 * 256 * 1024 + 255) / 256, B, 0, stream>>>(dec_tw1, p_tc1, 256, 256, 256);
  pack_tconv_w_kernel<<<(4 * 64 * 1024 + 255) / 256, B, 0, stream>>>(dec_tw2, p_tc2, 3, 64, 256);

  // --- encoder ---
  enc_conv1_kernel<<<(32L * 64 * 64 * 256) / 256, B, 0, stream>>>(x, enc_w1, enc_b1, big);
  gemm_conv_wmma_kernel<<<dim3(256, 4), B, 0, stream>>>(   // enc conv2 4x4 s2
      big, p_enc2, enc_b2, nullptr, t0, nullptr,
      64, 64, 256, 32, 32, 5, 5, 256, 256, 4, 2, 1, 4096, 0, 0, 0, 0);

  auto resblock = [&](u16* src, u16* tmp, u16* dst, int i) {
    gemm_conv_wmma_kernel<<<dim3(256, 4), B, 0, stream>>>(   // 3x3, relu on input
        src, p_rw1[i], rb1[i], nullptr, tmp, nullptr,
        32, 32, 256, 32, 32, 5, 5, 256, 256, 3, 1, 1, 2304, 0, 0, 1, 0);
    gemm_conv_wmma_kernel<<<dim3(256, 4), B, 0, stream>>>(   // 1x1, relu in, +residual
        tmp, p_rw2[i], rb2[i], src, dst, nullptr,
        32, 32, 256, 32, 32, 5, 5, 256, 256, 1, 1, 0, 256, 0, 0, 1, 0);
  };
  resblock(t0, t1, t2, 0);   // er1
  resblock(t2, t1, t0, 1);   // er2 -> z in t0
  nhwc_bf16_to_nchw_f32_kernel<<<(32L * 256 * 32 * 32) / 256, B, 0, stream>>>(t0, zout);

  // --- VQ: scores = z @ codes^T - 0.5||c||^2, argmax, gather ---
  gemm_conv_wmma_kernel<<<dim3(256, 8), B, 0, stream>>>(
      t0, p_codes, cn, nullptr, nullptr, scores,
      32, 32, 256, 32, 32, 5, 5, 512, 512, 1, 1, 0, 256, 0, 0, 0, 1);
  vq_argmin_gather_kernel<<<4096, B, 0, stream>>>(scores, codes, zdec, t1, 32, 32, 256, 512);

  // --- decoder (straight-through forward value == zdec, in t1) ---
  resblock(t1, t2, t0, 2);   // dr1
  resblock(t0, t2, t1, 3);   // dr2
  for (int cls = 0; cls < 4; ++cls)    // convT1: 32x32 -> 64x64 NHWC (big)
    gemm_conv_wmma_kernel<<<dim3(256, 4), B, 0, stream>>>(
        t1, p_tc1 + (long)cls * 256 * 1024, dec_tb1, nullptr, big, nullptr,
        32, 32, 256, 64, 64, 6, 6, 256, 256, 0, 0, 0, 1024, 1, cls, 0, 0);
  for (int cls = 0; cls < 4; ++cls)    // convT2: 64x64 -> (32,3,128,128) f32 NCHW
    gemm_conv_wmma_kernel<<<dim3(1024, 1), B, 0, stream>>>(
        big, p_tc2 + (long)cls * 64 * 1024, dec_tb2, nullptr, nullptr, xhat,
        64, 64, 256, 128, 128, 7, 7, 64, 3, 0, 0, 0, 1024, 1, cls, 0, 2);
}
